// IndRNN_77498389889321
// MI455X (gfx1250) — compile-verified, matrix operand order checked by reading, not run on hardware
//
#include <hip/hip_runtime.h>
#include <hip/hip_bf16.h>

// IndRNN fused kernel for gfx1250 (MI455X):
//   out[b,t,r] = h_t = relu(x[b,t,:]@W[:,r] + b[r] + u[r]*h_{t-1}),  h_0 = 1
// bf16 WMMA (f32 accum) for the projection GEMM, f32 for the recurrence.
// W^T resident in LDS (bf16); x streamed in 16-timestep chunks via CDNA5
// async global->LDS DMA (ASYNCcnt), double-staged (f32 staging + bf16 buffer)
// so HBM latency overlaps the WMMA+scan of the previous chunk.

typedef __attribute__((ext_vector_type(16))) __bf16 v16bf;
typedef __attribute__((ext_vector_type(8)))  float  v8f;

#define T_DIM 1024
#define I_DIM 512
#define R_DIM 512
#define B_DIM 64
#define NT    256          // R columns per workgroup
#define CHUNK 16           // timesteps per WMMA tile (M dimension)
#define NCHUNK (T_DIM / CHUNK)
#define WT_STRIDE 520      // 512 + 8 bf16 pad -> conflict-free ds_read_b128
#define X_STRIDE  520
#define WT_BYTES  (NT * WT_STRIDE * 2)          // 266,240
#define XBF_BYTES (CHUNK * X_STRIDE * 2)        //  16,640
#define XF32_BYTES (CHUNK * I_DIM * 4)          //  32,768 (linear staging)
#define SMEM_BYTES (WT_BYTES + XBF_BYTES + XF32_BYTES)   // 315,648 <= 320KB/WGP

__device__ __forceinline__ float relu_step(float acc, float bv, float uv, float h) {
    float p = acc + bv + uv * h;
    return p > 0.0f ? p : 0.0f;
}

// CDNA5 async DMA: 16B per lane, ASYNCcnt-tracked, no VGPR staging.
__device__ __forceinline__ void async_copy_b128(unsigned lds_off, const float* g) {
    asm volatile("global_load_async_to_lds_b128 %0, %1, off"
                 :: "v"(lds_off), "v"(g) : "memory");
}
__device__ __forceinline__ void wait_async0() {
    asm volatile("s_wait_asynccnt 0" ::: "memory");
}

// Issue one 32KB chunk (contiguous in global memory): 2048 x b128, 4 per thread.
__device__ __forceinline__ void issue_async_chunk(const float* src, unsigned lds_base, int tid) {
    #pragma unroll
    for (int j = 0; j < 4; ++j) {
        int e = tid + j * 512;                       // b128 index 0..2047
        async_copy_b128(lds_base + (unsigned)e * 16u, src + (size_t)e * 4);
    }
}

// LDS-local f32 -> bf16 (padded layout) conversion, 16 elems per thread.
__device__ __forceinline__ void convert_chunk(const float* lXf, __bf16* lX, int tid) {
    #pragma unroll
    for (int j = 0; j < 4; ++j) {
        int i = tid + j * 512;                       // float4 index 0..2047
        int e = i << 2;
        int m = e >> 9;                              // / 512
        int k = e & 511;
        const float4 v = *(const float4*)(lXf + e);
        __bf16* dst = &lX[m * X_STRIDE + k];
        dst[0] = (__bf16)v.x; dst[1] = (__bf16)v.y;
        dst[2] = (__bf16)v.z; dst[3] = (__bf16)v.w;
    }
}

__global__ __launch_bounds__(512, 1)
void indrnn_fused_kernel(const float* __restrict__ x,
                         const float* __restrict__ W,
                         const float* __restrict__ u,
                         const float* __restrict__ bias,
                         float* __restrict__ out) {
    extern __shared__ char smem[];
    __bf16* lWt = (__bf16*)smem;                       // [NT][WT_STRIDE]
    __bf16* lX  = (__bf16*)(smem + WT_BYTES);          // [CHUNK][X_STRIDE] bf16
    float*  lXf = (float*)(smem + WT_BYTES + XBF_BYTES); // [CHUNK*I_DIM] f32 staging

    const int tid    = threadIdx.x;
    const int wave   = tid >> 5;
    const int lane   = tid & 31;
    const int nloc16 = lane & 15;
    const bool hiH   = lane >= 16;
    const int batch  = blockIdx.y;
    const int rbase  = blockIdx.x * NT;

    const float* xb = x   + (size_t)batch * T_DIM * I_DIM;
    float*       ob = out + (size_t)batch * T_DIM * R_DIM;

    const unsigned ldsf_base = (unsigned)(size_t)(void*)lXf;  // wave-relative LDS byte addr

    // ---- kick off async DMA of chunk 0 immediately ----
    issue_async_chunk(xb, ldsf_base, tid);

    // ---- stage W^T into LDS as bf16 (coalesced over r); overlaps the async DMA ----
    for (int idx = tid; idx < NT * I_DIM; idx += 512) {
        int k = idx >> 8;            // / NT (=256)
        int n = idx & (NT - 1);
        lWt[n * WT_STRIDE + k] = (__bf16)W[k * R_DIM + rbase + n];
    }

    // per-lane channel constants (each lane owns column N = lane%16 of its wave tile)
    const int col  = rbase + wave * 16 + nloc16;
    const float uv = u[col];
    const float bv = bias[col];
    float hv = 1.0f;                 // h_0 = ones

    const int koff = hiH ? 8 : 0;    // 16-bit A/B fragment: hi half-lanes take K+8 / K+24
    const int mrow = nloc16;         // A row (timestep within chunk) for this lane

    // ---- prologue: chunk 0 -> bf16 buffer, then start DMA of chunk 1 ----
    wait_async0();
    __syncthreads();
    convert_chunk(lXf, lX, tid);
    __syncthreads();
    issue_async_chunk(xb + (size_t)1 * CHUNK * I_DIM, ldsf_base, tid);

    for (int c = 0; c < NCHUNK; ++c) {
        // ---- GEMM: 16(time) x 16(chan) tile, K = 512 via 16 chained bf16 WMMAs ----
        v8f acc = {};
        #pragma unroll
        for (int kk = 0; kk < I_DIM; kk += 32) {
            const int kb = kk + koff;
            v16bf afrag, bfrag;
            {   union { v16bf v; uint4 q[2]; } ua;
                ua.q[0] = *(const uint4*)&lX[mrow * X_STRIDE + kb];
                ua.q[1] = *(const uint4*)&lX[mrow * X_STRIDE + kb + 16];
                afrag = ua.v; }
            {   union { v16bf v; uint4 q[2]; } ub;
                ub.q[0] = *(const uint4*)&lWt[(wave * 16 + nloc16) * WT_STRIDE + kb];
                ub.q[1] = *(const uint4*)&lWt[(wave * 16 + nloc16) * WT_STRIDE + kb + 16];
                bfrag = ub.v; }
            acc = __builtin_amdgcn_wmma_f32_16x16x32_bf16(
                      false, afrag, false, bfrag, (short)0, acc, false, false);
        }

        // ---- sequential IndRNN scan over the 16 timesteps of this tile ----
        // C layout: lanes 0-15 hold M=0..7 (VGPR0..7), lanes 16-31 hold M=8..15.
        float h = hv;
        float res[8];
        if (!hiH) {
            #pragma unroll
            for (int v = 0; v < 8; ++v) { h = relu_step(acc[v], bv, uv, h); res[v] = h; }
        }
        float hmid = __shfl_xor(h, 16, 32);   // hi half receives h after step 7
        if (hiH) {
            h = hmid;
            #pragma unroll
            for (int v = 0; v < 8; ++v) { h = relu_step(acc[v], bv, uv, h); res[v] = h; }
        }
        float hfin = __shfl_xor(h, 16, 32);   // low half receives h after step 15
        hv = hiH ? h : hfin;

        // ---- store h (streaming output: non-temporal, keep L2 for x/W reuse) ----
        const int t0 = c * CHUNK + (hiH ? 8 : 0);
        #pragma unroll
        for (int v = 0; v < 8; ++v)
            __builtin_nontemporal_store(res[v], &ob[(size_t)(t0 + v) * R_DIM + col]);

        // ---- rotate pipeline: chunk c+1 f32 -> bf16, start DMA of chunk c+2 ----
        if (c < NCHUNK - 1) {
            wait_async0();                     // this wave's DMA of chunk c+1 done
            __syncthreads();                   // all waves: DMA visible + lX reads done
            convert_chunk(lXf, lX, tid);
            __syncthreads();                   // lX ready; lXf free to overwrite
            if (c < NCHUNK - 2)
                issue_async_chunk(xb + (size_t)(c + 2) * CHUNK * I_DIM, ldsf_base, tid);
        }
    }
}

extern "C" void kernel_launch(void* const* d_in, const int* in_sizes, int n_in,
                              void* d_out, int out_size, void* d_ws, size_t ws_size,
                              hipStream_t stream) {
    (void)in_sizes; (void)n_in; (void)out_size; (void)d_ws; (void)ws_size;
    const float* x  = (const float*)d_in[0];
    const float* W  = (const float*)d_in[1];
    const float* u  = (const float*)d_in[2];
    const float* b  = (const float*)d_in[3];
    float* out      = (float*)d_out;

    // 308 KB dynamic LDS (<= 320 KB/WGP on CDNA5); raise the HIP cap each call
    // (host-side, not a stream op -> graph-capture safe, deterministic).
    (void)hipFuncSetAttribute((const void*)indrnn_fused_kernel,
                              hipFuncAttributeMaxDynamicSharedMemorySize,
                              (int)SMEM_BYTES);

    dim3 grid(R_DIM / NT, B_DIM);   // 2 x 64 workgroups
    dim3 block(512);                // 16 wave32s
    indrnn_fused_kernel<<<grid, block, SMEM_BYTES, stream>>>(x, W, u, b, out);
}
